// DeeperDynamicEdgeNet_65171833750003
// MI455X (gfx1250) — compile-verified, matrix operand order checked by reading, not run on hardware
//
#include <hip/hip_runtime.h>
#include <cstddef>

// ---------------------------------------------------------------------------
// DeeperDynamicEdgeNet on MI455X (gfx1250, wave32, WMMA).
// Bandwidth-bound (~87 GFLOP, ~GBs of edge-activation traffic): keep all big
// intermediates in f16, matmuls via v_wmma_f32_16x16x32_f16 (fp32 acc),
// bias + BN statistics fused into GEMM epilogue, weight tiles staged to LDS
// with global_load_async_to_lds_b128 (ASYNCcnt path).
// ---------------------------------------------------------------------------

typedef _Float16 v16h __attribute__((ext_vector_type(16)));
typedef _Float16 v8h  __attribute__((ext_vector_type(8)));
typedef float    v8f  __attribute__((ext_vector_type(8)));

#define BGR   128           // graphs
#define NND   512           // nodes per graph
#define KNB   16            // neighbors
#define MEDGE (BGR*NND*KNB) // 1048576 edges
#define EPSB  1e-5f

// ============================== small helpers ==============================

__global__ void colstats3_kernel(const float* __restrict__ X, int M,
                                 float* __restrict__ stats /*[6]*/) {
  float s0=0,s1=0,s2=0,q0=0,q1=0,q2=0;
  for (int r = blockIdx.x*blockDim.x + threadIdx.x; r < M; r += gridDim.x*blockDim.x) {
    float a = X[(size_t)r*3+0], b = X[(size_t)r*3+1], c = X[(size_t)r*3+2];
    s0+=a; s1+=b; s2+=c; q0+=a*a; q1+=b*b; q2+=c*c;
  }
  __shared__ float ss[6];
  if (threadIdx.x < 6) ss[threadIdx.x] = 0.f;
  __syncthreads();
  atomicAdd(&ss[0],s0); atomicAdd(&ss[1],s1); atomicAdd(&ss[2],s2);
  atomicAdd(&ss[3],q0); atomicAdd(&ss[4],q1); atomicAdd(&ss[5],q2);
  __syncthreads();
  if (threadIdx.x < 6) atomicAdd(&stats[threadIdx.x], ss[threadIdx.x]);
}

__global__ void write_x1_kernel(const float* __restrict__ X, const float* __restrict__ stats,
                                const float* __restrict__ g, const float* __restrict__ be,
                                float* __restrict__ x1, int M) {
  int e = blockIdx.x*blockDim.x + threadIdx.x;
  if (e >= M*3) return;
  int c = e % 3;
  float invM = 1.f/(float)M;
  float m = stats[c]*invM;
  float v = stats[3+c]*invM - m*m;
  x1[e] = g[c]*(X[e]-m)*rsqrtf(v+EPSB) + be[c];
}

__global__ void f2h_pad_kernel(const float* __restrict__ F, int d,
                               _Float16* __restrict__ Fh, int Kp, int total) {
  int e = blockIdx.x*blockDim.x + threadIdx.x;
  if (e >= total) return;
  int c = e % Kp; int n = e / Kp;
  Fh[e] = (c < d) ? (_Float16)F[(size_t)n*d + c] : (_Float16)0.f;
}

__global__ void sqnorm_kernel(const float* __restrict__ F, int d,
                              float* __restrict__ sq, int total) {
  int t = blockIdx.x*blockDim.x + threadIdx.x;
  if (t >= total) return;
  const float* p = F + (size_t)t*d;
  float s = 0.f;
  for (int c = 0; c < d; ++c) s += p[c]*p[c];
  sq[t] = s;
}

// W[din,dout] fp32 -> WT[dout,Kp] f16 (transposed, zero K-pad)
__global__ void prep_wt_kernel(const float* __restrict__ W, int din, int dout,
                               _Float16* __restrict__ WT, int Kp) {
  int e = blockIdx.x*blockDim.x + threadIdx.x;
  if (e >= dout*Kp) return;
  int k = e % Kp, n = e / Kp;
  WT[e] = (k < din) ? (_Float16)W[(size_t)k*dout + n] : (_Float16)0.f;
}

// ====================== WMMA Gram matrix -> distances ======================
template<int NK>
__global__ __launch_bounds__(256) void gram_d2_kernel(const _Float16* __restrict__ Fh, int Kp,
                                                      const float* __restrict__ sq,
                                                      float* __restrict__ D2) {
  int b    = blockIdx.z;
  int wave = threadIdx.x >> 5, lane = threadIdx.x & 31;
  int i0   = blockIdx.x * 16;
  int j0   = (blockIdx.y * 8 + wave) * 16;
  int row  = lane & 15;
  int kb   = (lane < 16) ? 0 : 8;      // 16-bit A 16x32 layout: K split per half-wave
  const _Float16* Fb = Fh + (size_t)b * NND * Kp;
  const _Float16* ar = Fb + (size_t)(i0 + row) * Kp;
  const _Float16* br = Fb + (size_t)(j0 + row) * Kp;
  v8f c = {};
#pragma unroll
  for (int kk = 0; kk < NK; ++kk) {
    v16h a, bm;
    *(v8h*)&a      = *(const v8h*)(ar + kk*32 + kb);
    *((v8h*)&a+1)  = *(const v8h*)(ar + kk*32 + 16 + kb);
    *(v8h*)&bm     = *(const v8h*)(br + kk*32 + kb);
    *((v8h*)&bm+1) = *(const v8h*)(br + kk*32 + 16 + kb);
    c = __builtin_amdgcn_wmma_f32_16x16x32_f16(false, a, false, bm, (short)0, c, false, false);
  }
  int j = j0 + row;
  float sqj = sq[b*NND + j];
  int mbase = i0 + ((lane < 16) ? 0 : 8);
  float* drow = D2 + ((size_t)b*NND + mbase)*NND + j;
#pragma unroll
  for (int r = 0; r < 8; ++r)
    drow[(size_t)r*NND] = sq[b*NND + mbase + r] + sqj - 2.f*c[r];
}

// streaming top-16 smallest per row (self included, ascending like jax top_k(-d2))
__global__ void topk_kernel(const float* __restrict__ D2, int* __restrict__ idx) {
  int t = blockIdx.x*blockDim.x + threadIdx.x;
  if (t >= BGR*NND) return;
  const float4* row4 = (const float4*)(D2 + (size_t)t * NND);
  float bd[KNB]; int bj[KNB];
#pragma unroll
  for (int k = 0; k < KNB; ++k) { bd[k] = 3.4e38f; bj[k] = 0; }
  for (int j4 = 0; j4 < NND/4; ++j4) {
    float4 q = row4[j4];
    float vals[4] = {q.x, q.y, q.z, q.w};
#pragma unroll
    for (int s = 0; s < 4; ++s) {
      float d = vals[s];
      if (d < bd[KNB-1]) {
        int p = KNB-1;
        while (p > 0 && bd[p-1] > d) { bd[p] = bd[p-1]; bj[p] = bj[p-1]; --p; }
        bd[p] = d; bj[p] = j4*4 + s;
      }
    }
  }
  int* o = idx + (size_t)t*KNB;
#pragma unroll
  for (int k = 0; k < KNB; ++k) o[k] = bj[k];
}

// edge feature build: [xi, xj - xi] -> f16 K-padded rows; one wave per edge
__global__ __launch_bounds__(256) void build_edges_kernel(const float* __restrict__ F, int d,
                                                          const int* __restrict__ idx,
                                                          _Float16* __restrict__ A, int Kp) {
  int wave = threadIdx.x >> 5, lane = threadIdx.x & 31;
  long edge = (long)blockIdx.x * 8 + wave;
  long node = edge >> 4;                 // /K
  int  b    = (int)(node >> 9);          // /N
  int  j    = idx[edge];
  const float* xi = F + (size_t)node * d;
  const float* xj = F + ((size_t)b*NND + j) * d;
  _Float16* out = A + (size_t)edge * Kp;
  for (int c = lane; c < Kp; c += 32) {
    float v = 0.f;
    if (c < d)        v = xi[c];
    else if (c < 2*d) { int cc = c - d; v = xj[cc] - xi[cc]; }
    out[c] = (_Float16)v;
  }
}

// =================== edge MLP GEMM: Y = A*W + b, fused stats ===============
// Weight tile (16 rows x Kp, contiguous in WT) staged once per block into LDS
// via async global->LDS (ASYNCcnt), B-fragments then read with ds_load_b128.
template<int NK>
__global__ __launch_bounds__(256) void edge_gemm_kernel(const _Float16* __restrict__ A, int Kp,
                                                        const _Float16* __restrict__ WT,
                                                        const float* __restrict__ bias,
                                                        _Float16* __restrict__ Y, int Nout,
                                                        float* __restrict__ statsSum,
                                                        float* __restrict__ statsSq) {
  __shared__ _Float16 sWT[16*160];
  __shared__ float s_sum[16], s_sq[16];
  int wave = threadIdx.x >> 5, lane = threadIdx.x & 31;
  if (threadIdx.x < 16) { s_sum[threadIdx.x] = 0.f; s_sq[threadIdx.x] = 0.f; }
  int n0 = blockIdx.y * 16;
  if (wave == 0) {
    // async copy of the 16 x Kp f16 weight tile (rows n0..n0+15 are contiguous)
    const char* gsrc = (const char*)(WT + (size_t)n0 * Kp);
    char* lbase = (char*)&sWT[0];
    int nchunks = (16 * Kp * 2) >> 4;          // 16B chunks
    for (int t = lane; t < nchunks; t += 32) {
      unsigned lds = (unsigned)(size_t)(lbase + t*16);
      const char* ga = gsrc + (size_t)t*16;
      asm volatile("global_load_async_to_lds_b128 %0, %1, off"
                   :: "v"(lds), "v"(ga) : "memory");
    }
    asm volatile("s_wait_asynccnt 0x0" ::: "memory");
  }
  __syncthreads();

  long mBase = ((long)blockIdx.x * 8 + wave) * 64;   // 4 m-tiles of 16 per wave
  int  row   = lane & 15;
  int  kb    = (lane < 16) ? 0 : 8;
  v16h bfrag[NK];
  const _Float16* wrow = sWT + (size_t)row * Kp;     // LDS reads -> ds_load_b128
#pragma unroll
  for (int kk = 0; kk < NK; ++kk) {
    v16h t;
    *(v8h*)&t     = *(const v8h*)(wrow + kk*32 + kb);
    *((v8h*)&t+1) = *(const v8h*)(wrow + kk*32 + 16 + kb);
    bfrag[kk] = t;
  }
  float bval = bias[n0 + row];
  float lsum = 0.f, lsq = 0.f;
#pragma unroll
  for (int mt = 0; mt < 4; ++mt) {
    long m0 = mBase + mt*16;
    const _Float16* ar = A + (size_t)(m0 + row) * Kp;
    __builtin_prefetch(ar + 16*Kp, 0, 1);            // -> global_prefetch_b8
    v8f c = {};
#pragma unroll
    for (int kk = 0; kk < NK; ++kk) {
      v16h a;
      *(v8h*)&a     = *(const v8h*)(ar + kk*32 + kb);
      *((v8h*)&a+1) = *(const v8h*)(ar + kk*32 + 16 + kb);
      c = __builtin_amdgcn_wmma_f32_16x16x32_f16(false, a, false, bfrag[kk], (short)0, c, false, false);
    }
    long mrow = m0 + ((lane < 16) ? 0 : 8);
    _Float16* yb = Y + (size_t)mrow * Nout + n0 + row;
#pragma unroll
    for (int r = 0; r < 8; ++r) {
      float v = c[r] + bval;
      lsum += v; lsq += v*v;
      yb[(size_t)r * Nout] = (_Float16)v;
    }
  }
  atomicAdd(&s_sum[row], lsum);
  atomicAdd(&s_sq[row],  lsq);
  __syncthreads();
  if (threadIdx.x < 16) {
    atomicAdd(&statsSum[n0 + threadIdx.x], s_sum[threadIdx.x]);
    atomicAdd(&statsSq[n0 + threadIdx.x],  s_sq[threadIdx.x]);
  }
}

// elementwise BN (batch stats, biased var) + ReLU, in place, 8 halfs per thread
__global__ void bn_relu_kernel(_Float16* __restrict__ Y, int Nout, long M,
                               const float* __restrict__ sum, const float* __restrict__ sumsq,
                               const float* __restrict__ g, const float* __restrict__ be) {
  long e = ((long)blockIdx.x * blockDim.x + threadIdx.x) * 8;
  if (e >= M * Nout) return;
  int c0 = (int)(e % Nout);
  v8h y = *(v8h*)(Y + e);
  v8f sm = *(const v8f*)(sum + c0);
  v8f sv = *(const v8f*)(sumsq + c0);
  v8f gg = *(const v8f*)(g + c0);
  v8f bb = *(const v8f*)(be + c0);
  float invM = 1.f/(float)M;
  v8h o;
#pragma unroll
  for (int t = 0; t < 8; ++t) {
    float m = sm[t]*invM;
    float v = sv[t]*invM - m*m;
    float val = gg[t]*((float)y[t] - m)*rsqrtf(v+EPSB) + bb[t];
    o[t] = (_Float16)fmaxf(val, 0.f);
  }
  *(v8h*)(Y + e) = o;
}

// mean over K neighbors + concat with x1 -> fp32 node features [B,N, 3+dout]
__global__ void mean_concat_kernel(const _Float16* __restrict__ Yf, int dout,
                                   const float* __restrict__ x1,
                                   float* __restrict__ Fn, int dn, int total) {
  int e = blockIdx.x*blockDim.x + threadIdx.x;
  if (e >= total) return;
  int  f    = e % dn;
  long node = e / dn;
  if (f < 3) { Fn[e] = x1[node*3 + f]; return; }
  const _Float16* p = Yf + (size_t)(node*KNB) * dout + (f - 3);
  float s = 0.f;
#pragma unroll
  for (int kk = 0; kk < KNB; ++kk) s += (float)p[(size_t)kk*dout];
  Fn[e] = s * (1.f/(float)KNB);
}

// ============================== readout head ===============================

__global__ void bn_u_kernel(const float* __restrict__ u, const float* __restrict__ g,
                            const float* __restrict__ be, float* __restrict__ ghead) {
  int c = threadIdx.x;
  if (c >= 2) return;
  float s = 0.f, q = 0.f;
  for (int b = 0; b < BGR; ++b) { float v = u[b*2+c]; s += v; q += v*v; }
  float m = s/(float)BGR, var = q/(float)BGR - m*m;
  float sc = g[c]*rsqrtf(var+EPSB);
  for (int b = 0; b < BGR; ++b) ghead[b*133 + c] = (u[b*2+c]-m)*sc + be[c];
}

__global__ void u2_kernel(const float* __restrict__ Ffin, float* __restrict__ ghead) {
  int b = blockIdx.x;
  for (int f = threadIdx.x; f < 131; f += blockDim.x) {
    float s = 0.f;
    for (int n = 0; n < NND; ++n) s += Ffin[((size_t)b*NND + n)*131 + f];
    ghead[b*133 + 2 + f] = s * (1.f/(float)NND);
  }
}

__global__ __launch_bounds__(128) void head1_kernel(const float* __restrict__ ghead,
                                                    const float* __restrict__ W1,
                                                    const float* __restrict__ b1,
                                                    const float* __restrict__ g1,
                                                    const float* __restrict__ be1,
                                                    float* __restrict__ Hh) {
  int f = blockIdx.x, b = threadIdx.x;
  float acc = b1[f];
  for (int k = 0; k < 133; ++k) acc += ghead[b*133 + k] * W1[(size_t)k*256 + f];
  __shared__ float ss[128], sq[128];
  ss[b] = acc; sq[b] = acc*acc;
  __syncthreads();
  for (int s = 64; s > 0; s >>= 1) {
    if (b < s) { ss[b] += ss[b+s]; sq[b] += sq[b+s]; }
    __syncthreads();
  }
  float m = ss[0]/(float)BGR;
  float v = sq[0]/(float)BGR - m*m;
  float val = g1[f]*(acc-m)*rsqrtf(v+EPSB) + be1[f];
  Hh[b*256 + f] = fmaxf(val, 0.f);
}

__global__ void head2_kernel(const float* __restrict__ Hh, const float* __restrict__ W2,
                             const float* __restrict__ b2, float* __restrict__ out) {
  int b = threadIdx.x;
  if (b >= BGR) return;
  float acc = b2[0];
  for (int f = 0; f < 256; ++f) acc += Hh[b*256 + f] * W2[f];
  out[b] = acc;
}

// ================================ launch ===================================

extern "C" void kernel_launch(void* const* d_in, const int* in_sizes, int n_in,
                              void* d_out, int out_size, void* d_ws, size_t ws_size,
                              hipStream_t stream) {
  (void)in_sizes; (void)n_in; (void)out_size; (void)ws_size;

  const float* x = (const float*)d_in[0];
  const float* u = (const float*)d_in[1];
  struct Lay { int base; int din; int Kp; int dout; };
  const Lay layers[9] = {
    { 2,   6,  32,  32}, { 6,  32,  32,  32}, {10,  32,  32,  32},   // conv1
    {14,  70,  96,  64}, {18,  64,  64,  64}, {22,  64,  64,  64},   // conv2
    {26, 134, 160, 128}, {30, 128, 128, 128}, {34, 128, 128, 128}};  // conv3
  const float* bnx_g  = (const float*)d_in[38];
  const float* bnx_be = (const float*)d_in[39];
  const float* bnu_g  = (const float*)d_in[40];
  const float* bnu_be = (const float*)d_in[41];
  const float* W1h    = (const float*)d_in[42];
  const float* b1h    = (const float*)d_in[43];
  const float* g1h    = (const float*)d_in[44];
  const float* be1h   = (const float*)d_in[45];
  const float* W2h    = (const float*)d_in[46];
  const float* b2h    = (const float*)d_in[47];

  size_t off = 0;
  auto alloc = [&](size_t bytes) { off = (off + 255) & ~(size_t)255; size_t o = off; off += bytes; return o; };
  char* ws = (char*)d_ws;
  const int BN = BGR*NND;
  _Float16* EBUF  = (_Float16*)(ws + alloc((size_t)MEDGE*160*2));
  _Float16* YB1   = (_Float16*)(ws + alloc((size_t)MEDGE*128*2));
  _Float16* YB2   = (_Float16*)(ws + alloc((size_t)MEDGE*128*2));
  float*    D2    = (float*)   (ws + alloc((size_t)BGR*NND*NND*4));
  _Float16* Fh    = (_Float16*)(ws + alloc((size_t)BN*96*2));
  float*    F2    = (float*)   (ws + alloc((size_t)BN*35*4));
  float*    F3    = (float*)   (ws + alloc((size_t)BN*67*4));
  float*    Ffin  = (float*)   (ws + alloc((size_t)BN*131*4));
  float*    x1    = (float*)   (ws + alloc((size_t)BN*3*4));
  float*    sq    = (float*)   (ws + alloc((size_t)BN*4));
  int*      idx   = (int*)     (ws + alloc((size_t)MEDGE*4));
  float*    stats = (float*)   (ws + alloc(512*4));
  _Float16* WT    = (_Float16*)(ws + alloc((size_t)9*20480*2));
  float*    ghead = (float*)   (ws + alloc((size_t)BGR*133*4));
  float*    Hh    = (float*)   (ws + alloc((size_t)BGR*256*4));
  float* statsSum = stats;
  float* statsSq  = stats + 256;

  for (int l = 0; l < 9; ++l) {
    int n = layers[l].dout * layers[l].Kp;
    prep_wt_kernel<<<(n+255)/256, 256, 0, stream>>>((const float*)d_in[layers[l].base],
                                                    layers[l].din, layers[l].dout,
                                                    WT + (size_t)l*20480, layers[l].Kp);
  }

  hipMemsetAsync(stats, 0, 512*sizeof(float), stream);
  colstats3_kernel<<<256, 256, 0, stream>>>(x, BN, stats);
  write_x1_kernel<<<(BN*3+255)/256, 256, 0, stream>>>(x, stats, bnx_g, bnx_be, x1, BN);

  auto launch_gram = [&](int Kp) {
    dim3 g(NND/16, NND/16/8, BGR);
    switch (Kp >> 5) {
      case 1: gram_d2_kernel<1><<<g, 256, 0, stream>>>(Fh, Kp, sq, D2); break;
      case 2: gram_d2_kernel<2><<<g, 256, 0, stream>>>(Fh, Kp, sq, D2); break;
      default: gram_d2_kernel<3><<<g, 256, 0, stream>>>(Fh, Kp, sq, D2); break;
    }
  };
  auto launch_gemm = [&](const _Float16* A, int Kp, const _Float16* Wt, const float* bias,
                         _Float16* Y, int Nout, const float* g, const float* be) {
    hipMemsetAsync(stats, 0, 512*sizeof(float), stream);
    dim3 gd(MEDGE/512, Nout/16);
    switch (Kp >> 5) {
      case 1: edge_gemm_kernel<1><<<gd, 256, 0, stream>>>(A, Kp, Wt, bias, Y, Nout, statsSum, statsSq); break;
      case 2: edge_gemm_kernel<2><<<gd, 256, 0, stream>>>(A, Kp, Wt, bias, Y, Nout, statsSum, statsSq); break;
      case 3: edge_gemm_kernel<3><<<gd, 256, 0, stream>>>(A, Kp, Wt, bias, Y, Nout, statsSum, statsSq); break;
      case 4: edge_gemm_kernel<4><<<gd, 256, 0, stream>>>(A, Kp, Wt, bias, Y, Nout, statsSum, statsSq); break;
      default: edge_gemm_kernel<5><<<gd, 256, 0, stream>>>(A, Kp, Wt, bias, Y, Nout, statsSum, statsSq); break;
    }
    long tot = (long)MEDGE * Nout;
    bn_relu_kernel<<<(unsigned)(tot/(8*256)), 256, 0, stream>>>(Y, Nout, MEDGE, statsSum, statsSq, g, be);
  };

  auto edge_conv = [&](const float* Fin, int d, int KpF, int KpE, int lbase,
                       float* Fout, int dnOut) {
    int tFh = BN * KpF;
    f2h_pad_kernel<<<(tFh+255)/256, 256, 0, stream>>>(Fin, d, Fh, KpF, tFh);
    sqnorm_kernel<<<(BN+255)/256, 256, 0, stream>>>(Fin, d, sq, BN);
    launch_gram(KpF);
    topk_kernel<<<(BGR*NND+255)/256, 256, 0, stream>>>(D2, idx);
    build_edges_kernel<<<MEDGE/8, 256, 0, stream>>>(Fin, d, idx, EBUF, KpE);
    const Lay& L0 = layers[lbase+0]; const Lay& L1 = layers[lbase+1]; const Lay& L2 = layers[lbase+2];
    launch_gemm(EBUF, L0.Kp, WT + (size_t)(lbase+0)*20480, (const float*)d_in[L0.base+1],
                YB1, L0.dout, (const float*)d_in[L0.base+2], (const float*)d_in[L0.base+3]);
    launch_gemm(YB1, L1.Kp, WT + (size_t)(lbase+1)*20480, (const float*)d_in[L1.base+1],
                YB2, L1.dout, (const float*)d_in[L1.base+2], (const float*)d_in[L1.base+3]);
    launch_gemm(YB2, L2.Kp, WT + (size_t)(lbase+2)*20480, (const float*)d_in[L2.base+1],
                YB1, L2.dout, (const float*)d_in[L2.base+2], (const float*)d_in[L2.base+3]);
    int tot = BN * dnOut;
    mean_concat_kernel<<<(tot+255)/256, 256, 0, stream>>>(YB1, L2.dout, x1, Fout, dnOut, tot);
  };

  edge_conv(x,  3,  32,  32, 0, F2,   35);   // conv1 (kNN & edges on raw x)
  edge_conv(F2, 35, 64,  96, 3, F3,   67);   // conv2
  edge_conv(F3, 67, 96, 160, 6, Ffin, 131);  // conv3

  bn_u_kernel<<<1, 32, 0, stream>>>(u, bnu_g, bnu_be, ghead);
  u2_kernel<<<BGR, 128, 0, stream>>>(Ffin, ghead);
  head1_kernel<<<256, 128, 0, stream>>>(ghead, W1h, b1h, g1h, be1h, Hh);
  head2_kernel<<<1, 128, 0, stream>>>(Hh, W2h, b2h, (float*)d_out);
}
